// Attention_16947940950618
// MI455X (gfx1250) — matrix operand order repaired and checked by reading, not compiled, not run on hardware
//
#include <hip/hip_runtime.h>
#include <hip/hip_bf16.h>

// ---------------- problem constants ----------------
#define B_    2
#define N_    2048
#define DIM_  1024
#define H_    16
#define DH_   64
#define NK_   2
#define HD_   1024            // H_*DH_
#define KVROWS 2080           // NK_+N_ (=2050) padded up so 32-key tiles never read OOB
#define SCALE_ 0.125f         // DH^-0.5
#define NEGBIG (-3.4028234663852886e+38f)

typedef __attribute__((ext_vector_type(16))) __bf16 v16bf;
typedef __attribute__((ext_vector_type(8)))  float  v8f;
typedef __attribute__((ext_vector_type(4)))  unsigned int u32x4;

union Frag { v16bf v; u32x4 q[2]; };

__device__ __forceinline__ unsigned short f2bf(float x) {
  unsigned int u = __float_as_uint(x);
  unsigned int r = u + 0x7FFFu + ((u >> 16) & 1u);   // round-to-nearest-even
  return (unsigned short)(r >> 16);
}

// A fragment (16x32 bf16, M x K): lane L holds row m=L&15; elements e<8 at K=kb+8*hf+e,
// e>=8 at K=kb+16+8*hf+(e-8). Two 16B contiguous chunks.
__device__ __forceinline__ v16bf frag_a(const unsigned short* row, int kb, int hf) {
  Frag f;
  f.q[0] = *(const u32x4*)(row + kb + hf * 8);
  f.q[1] = *(const u32x4*)(row + kb + 16 + hf * 8);
  return f.v;
}

// B fragment (32x16 bf16, K x N) read from a K-contiguous "column" pointer
// (i.e. transposed storage): lane L is column n=L&15; elements e<8 at K=kb+8*hf+e,
// e>=8 at K=kb+16+8*hf+(e-8).
__device__ __forceinline__ v16bf frag_b(const unsigned short* col, int kb, int hf) {
  Frag f;
  f.q[0] = *(const u32x4*)(col + kb + hf * 8);
  f.q[1] = *(const u32x4*)(col + kb + 16 + hf * 8);
  return f.v;
}

__device__ __forceinline__ v8f wmma_bf16(v16bf a, v16bf b, v8f c) {
  return __builtin_amdgcn_wmma_f32_16x16x32_bf16(false, a, false, b, (short)0, c, false, false);
}

// ---------------- weight cast + transpose: dst[n][k] = (bf16)src[k][n] ----------------
__global__ void transpose_cast_kernel(const float* __restrict__ src,
                                      unsigned short* __restrict__ dst,
                                      int K, int Nout) {
  int idx = blockIdx.x * blockDim.x + threadIdx.x;
  if (idx >= K * Nout) return;
  int k = idx / Nout, n = idx % Nout;
  dst[(size_t)n * K + k] = f2bf(src[idx]);
}

// ---------------- layernorm + bf16 casts ----------------
__global__ void layernorm_cast_kernel(const float* __restrict__ x,
                                      const float* __restrict__ gamma,
                                      unsigned short* __restrict__ xn,
                                      unsigned short* __restrict__ xb) {
  __shared__ float ssum[256];
  __shared__ float ssq[256];
  const int row = blockIdx.x;          // b*N + n
  const int tid = threadIdx.x;
  const float4* xr4 = (const float4*)(x + (size_t)row * DIM_);
  float4 t4 = xr4[tid];
  float v0 = t4.x, v1 = t4.y, v2 = t4.z, v3 = t4.w;
  float s = v0 + v1 + v2 + v3;
  float sq = v0*v0 + v1*v1 + v2*v2 + v3*v3;
  ssum[tid] = s; ssq[tid] = sq;
  __syncthreads();
  for (int stride = 128; stride > 0; stride >>= 1) {
    if (tid < stride) { ssum[tid] += ssum[tid + stride]; ssq[tid] += ssq[tid + stride]; }
    __syncthreads();
  }
  const float mean = ssum[0] * (1.0f / DIM_);
  const float var  = ssq[0] * (1.0f / DIM_) - mean * mean;
  const float rstd = rsqrtf(var + 1e-5f);
  const int c0 = tid * 4;
  unsigned short* xnr = xn + (size_t)row * DIM_;
  unsigned short* xbr = xb + (size_t)row * DIM_;
  float g0 = gamma[c0], g1 = gamma[c0+1], g2 = gamma[c0+2], g3 = gamma[c0+3];
  xnr[c0  ] = f2bf((v0 - mean) * rstd * g0);
  xnr[c0+1] = f2bf((v1 - mean) * rstd * g1);
  xnr[c0+2] = f2bf((v2 - mean) * rstd * g2);
  xnr[c0+3] = f2bf((v3 - mean) * rstd * g3);
  xbr[c0  ] = f2bf(v0);
  xbr[c0+1] = f2bf(v1);
  xbr[c0+2] = f2bf(v2);
  xbr[c0+3] = f2bf(v3);
}

// ---------------- WMMA GEMM mainloop: per-wave 16(M) x 64(N) tile ----------------
// Register double-buffered: K-step k+1 fragments are loaded while step k's WMMAs run,
// so the scheduler can overlap global_load_b128 clauses with v_wmma instead of
// draining loadcnt to 0 before every WMMA.
__device__ __forceinline__ void gemm_tile_16x64(const unsigned short* __restrict__ A,
                                                const unsigned short* __restrict__ Bt,
                                                int K, int row0, int col0, v8f acc[4]) {
  const int lane = threadIdx.x & 31;
  const int hf = lane >> 4;
  const int lm = lane & 15;
  const unsigned short* arow = A + (size_t)(row0 + lm) * K;
  const unsigned short* b0 = Bt + (size_t)(col0 + lm) * K;
  const unsigned short* b1 = b0 + (size_t)16 * K;
  const unsigned short* b2 = b0 + (size_t)32 * K;
  const unsigned short* b3 = b0 + (size_t)48 * K;
  v16bf af  = frag_a(arow, 0, hf);
  v16bf bf0 = frag_b(b0, 0, hf);
  v16bf bf1 = frag_b(b1, 0, hf);
  v16bf bf2 = frag_b(b2, 0, hf);
  v16bf bf3 = frag_b(b3, 0, hf);
  for (int kb = 0; kb < K; kb += 32) {
    // branchless prefetch (last iteration re-reads the same in-bounds tile)
    const int kn = (kb + 32 < K) ? (kb + 32) : kb;
    v16bf afn  = frag_a(arow, kn, hf);
    v16bf bf0n = frag_b(b0, kn, hf);
    v16bf bf1n = frag_b(b1, kn, hf);
    v16bf bf2n = frag_b(b2, kn, hf);
    v16bf bf3n = frag_b(b3, kn, hf);
    acc[0] = wmma_bf16(af, bf0, acc[0]);
    acc[1] = wmma_bf16(af, bf1, acc[1]);
    acc[2] = wmma_bf16(af, bf2, acc[2]);
    acc[3] = wmma_bf16(af, bf3, acc[3]);
    af = afn; bf0 = bf0n; bf1 = bf1n; bf2 = bf2n; bf3 = bf3n;
  }
}

#define GEMM_PROLOG \
  const int lane = threadIdx.x & 31;                     \
  const int w = threadIdx.x >> 5;                        \
  const int hf = lane >> 4;                              \
  const int lm = lane & 15;                              \
  const int row0 = blockIdx.y * 128 + w * 16;            \
  const int col0 = blockIdx.x * 64;                      \
  v8f acc[4];                                            \
  v8f zz = {0.f,0.f,0.f,0.f,0.f,0.f,0.f,0.f};            \
  acc[0] = zz; acc[1] = zz; acc[2] = zz; acc[3] = zz;

// Q = xn @ Wq, scaled, scattered to [b][h][n][d] bf16
__global__ void gemm_q_kernel(const unsigned short* __restrict__ A,
                              const unsigned short* __restrict__ Bt,
                              unsigned short* __restrict__ qout) {
  GEMM_PROLOG
  gemm_tile_16x64(A, Bt, DIM_, row0, col0, acc);
  #pragma unroll
  for (int g = 0; g < 4; g++) {
    int nout = col0 + g * 16 + lm;
    int h = nout >> 6, d = nout & 63;
    #pragma unroll
    for (int vv = 0; vv < 8; vv++) {
      int grow = row0 + vv + 8 * hf;
      int b = grow >> 11, nrow = grow & (N_ - 1);
      qout[(((size_t)(b * H_ + h) * N_) + nrow) * DH_ + d] = f2bf(acc[g][vv] * SCALE_);
    }
  }
}

// KV = x @ Wkv -> K row-major [b][NK+n][d], V transposed [b][d][NK+n]
__global__ void gemm_kv_kernel(const unsigned short* __restrict__ A,
                               const unsigned short* __restrict__ Bt,
                               unsigned short* __restrict__ Kbuf,
                               unsigned short* __restrict__ Vt) {
  GEMM_PROLOG
  gemm_tile_16x64(A, Bt, DIM_, row0, col0, acc);
  #pragma unroll
  for (int g = 0; g < 4; g++) {
    int nout = col0 + g * 16 + lm;        // 0..127
    #pragma unroll
    for (int vv = 0; vv < 8; vv++) {
      int grow = row0 + vv + 8 * hf;
      int b = grow >> 11, nrow = grow & (N_ - 1);
      unsigned short val = f2bf(acc[g][vv]);
      if (nout < DH_)
        Kbuf[((size_t)b * KVROWS + NK_ + nrow) * DH_ + nout] = val;
      else
        Vt[((size_t)b * DH_ + (nout - DH_)) * KVROWS + NK_ + nrow] = val;
    }
  }
}

// out = attn_out @ Wout -> f32
__global__ void gemm_out_kernel(const unsigned short* __restrict__ A,
                                const unsigned short* __restrict__ Bt,
                                float* __restrict__ out) {
  GEMM_PROLOG
  gemm_tile_16x64(A, Bt, HD_, row0, col0, acc);
  #pragma unroll
  for (int g = 0; g < 4; g++) {
    int nout = col0 + g * 16 + lm;
    #pragma unroll
    for (int vv = 0; vv < 8; vv++) {
      int grow = row0 + vv + 8 * hf;
      out[(size_t)grow * DIM_ + nout] = acc[g][vv];
    }
  }
}

// ---------------- null-KV fill + zero key/value padding rows ----------------
__global__ void kv_init_kernel(const float* __restrict__ null_kv,
                               unsigned short* __restrict__ Kbuf,
                               unsigned short* __restrict__ Vt) {
  int t = blockIdx.x * blockDim.x + threadIdx.x;      // B_*64*32
  if (t >= B_ * DH_ * 32) return;
  int b = t / (DH_ * 32);
  int r = t % (DH_ * 32);
  int d = r / 32;
  int slot = r % 32;                                   // 0..1 null, 2..31 padding
  int row = (slot < NK_) ? slot : (NK_ + N_ + (slot - NK_));
  float kvalue = (slot < NK_) ? null_kv[slot * DH_ + d] : 0.0f;           // null_kv[0]
  float vvalue = (slot < NK_) ? null_kv[(NK_ + slot) * DH_ + d] : 0.0f;   // null_kv[1]
  Kbuf[((size_t)b * KVROWS + row) * DH_ + d] = f2bf(kvalue);
  Vt[((size_t)b * DH_ + d) * KVROWS + row]   = f2bf(vvalue);
}

// ---------------- flash attention: per-wave 16-query tile ----------------
__global__ void attention_kernel(const unsigned short* __restrict__ qbuf,
                                 const unsigned short* __restrict__ Kbuf,
                                 const unsigned short* __restrict__ Vt,
                                 const float* __restrict__ bias,
                                 const int* __restrict__ mask,
                                 unsigned short* __restrict__ attn_out) {
  __shared__ __align__(16) unsigned short Plds[8][16][32];   // per-wave 16x32 bf16
  const int b = blockIdx.z, h = blockIdx.y;
  const int w = threadIdx.x >> 5;
  const int lane = threadIdx.x & 31;
  const int hf = lane >> 4;
  const int lm = lane & 15;
  const int q0 = blockIdx.x * 128 + w * 16;

  // Q fragments for this wave's 16 rows (K-dim = DH = 64 -> two 32-chunks)
  const unsigned short* qrow = qbuf + (((size_t)(b * H_ + h) * N_) + q0 + lm) * DH_;
  v16bf qf0 = frag_a(qrow, 0, hf);
  v16bf qf1 = frag_a(qrow, 32, hf);

  v8f zz = {0.f,0.f,0.f,0.f,0.f,0.f,0.f,0.f};
  v8f o[4]; o[0] = zz; o[1] = zz; o[2] = zz; o[3] = zz;
  float mrow[8], lrow[8];
  #pragma unroll
  for (int v = 0; v < 8; v++) { mrow[v] = NEGBIG; lrow[v] = 0.0f; }

  const int jend = q0 + 15 + NK_ + 1;        // exclusive bound on allowed j
  const int ntile = (jend + 31) >> 5;

  for (int jt = 0; jt < ntile; jt++) {
    const int j0 = jt * 32;

    // ---- issue ALL of this tile's independent loads up front ----
    // K fragments (2 sub-tiles x 2 K-chunks)
    const unsigned short* krow0 = Kbuf + ((size_t)b * KVROWS + j0 + lm) * DH_;
    const unsigned short* krow1 = krow0 + (size_t)16 * DH_;
    v16bf k00 = frag_b(krow0, 0, hf);
    v16bf k01 = frag_b(krow0, 32, hf);
    v16bf k10 = frag_b(krow1, 0, hf);
    v16bf k11 = frag_b(krow1, 32, hf);
    // V fragments (consumed only after the LDS round-trip -> long overlap window)
    const unsigned short* vcol = Vt + ((size_t)b * DH_ + lm) * KVROWS + j0;
    v16bf vf0 = frag_b(vcol, 0, hf);
    v16bf vf1 = frag_b(vcol + (size_t)16 * KVROWS, 0, hf);
    v16bf vf2 = frag_b(vcol + (size_t)32 * KVROWS, 0, hf);
    v16bf vf3 = frag_b(vcol + (size_t)48 * KVROWS, 0, hf);
    // key-padding mask + bias tile (the dominant HBM stream) -- no dep on WMMA
    bool mok[2];
    float btile[2][8];
    #pragma unroll
    for (int kk = 0; kk < 2; kk++) {
      const int j = j0 + kk * 16 + lm;
      const bool jvalid = (j >= NK_) && (j < NK_ + N_);
      mok[kk] = (j < NK_) || (jvalid && mask[b * N_ + (j - NK_)] != 0);
      #pragma unroll
      for (int v = 0; v < 8; v++) {
        const int i = q0 + v + 8 * hf;
        btile[kk][v] = jvalid ? bias[((size_t)h * N_ + i) * N_ + (j - NK_)] : 0.0f;
      }
    }

    // ---- S = Q @ K^T over 32 keys ----
    v8f s[2]; s[0] = zz; s[1] = zz;
    s[0] = wmma_bf16(qf0, k00, s[0]);
    s[0] = wmma_bf16(qf1, k01, s[0]);
    s[1] = wmma_bf16(qf0, k10, s[1]);
    s[1] = wmma_bf16(qf1, k11, s[1]);

    // ---- bias + mask + online softmax ----
    float p[2][8];
    #pragma unroll
    for (int v = 0; v < 8; v++) {
      const int i = q0 + v + 8 * hf;
      float best = mrow[v];
      #pragma unroll
      for (int kk = 0; kk < 2; kk++) {
        const int j = j0 + kk * 16 + lm;
        const bool allowed = mok[kk] && (j <= i + NK_);
        float val = allowed ? (s[kk][v] + btile[kk][v]) : NEGBIG;
        p[kk][v] = val;
        best = fmaxf(best, val);
      }
      #pragma unroll
      for (int off = 1; off < 16; off <<= 1) best = fmaxf(best, __shfl_xor(best, off, 32));
      const float alpha = __expf(mrow[v] - best);
      mrow[v] = best;
      float lsum = 0.0f;
      #pragma unroll
      for (int kk = 0; kk < 2; kk++) {
        float e = __expf(p[kk][v] - best);
        p[kk][v] = e;
        lsum += e;
      }
      #pragma unroll
      for (int off = 1; off < 16; off <<= 1) lsum += __shfl_xor(lsum, off, 32);
      lrow[v] = lrow[v] * alpha + lsum;
      o[0][v] *= alpha; o[1][v] *= alpha; o[2][v] *= alpha; o[3][v] *= alpha;
    }

    // ---- P: D-layout -> A-layout via per-wave LDS round trip ----
    #pragma unroll
    for (int v = 0; v < 8; v++) {
      Plds[w][v + 8 * hf][lm]      = f2bf(p[0][v]);
      Plds[w][v + 8 * hf][16 + lm] = f2bf(p[1][v]);
    }
    asm volatile("s_wait_dscnt 0" ::: "memory");   // per-wave LDS RAW fence
    Frag pf;
    const unsigned short* prow = &Plds[w][lm][0];
    pf.q[0] = *(const u32x4*)(prow + hf * 8);
    pf.q[1] = *(const u32x4*)(prow + 16 + hf * 8);

    // ---- O += P @ V (V stored transposed: [b][d][kvrow]) ----
    o[0] = wmma_bf16(pf.v, vf0, o[0]);
    o[1] = wmma_bf16(pf.v, vf1, o[1]);
    o[2] = wmma_bf16(pf.v, vf2, o[2]);
    o[3] = wmma_bf16(pf.v, vf3, o[3]);
  }

  // ---- finalize: O / l, store bf16 [b][n][h*64+d] ----
  #pragma unroll
  for (int v = 0; v < 8; v++) {
    const int i = q0 + v + 8 * hf;
    const float inv = 1.0f / lrow[v];
    #pragma unroll
    for (int g = 0; g < 4; g++) {
      const int d = g * 16 + lm;
      attn_out[((size_t)(b * N_) + i) * HD_ + h * DH_ + d] = f2bf(o[g][v] * inv);
    }
  }
}

// ---------------- host launcher ----------------
extern "C" void kernel_launch(void* const* d_in, const int* in_sizes, int n_in,
                              void* d_out, int out_size, void* d_ws, size_t ws_size,
                              hipStream_t stream) {
  (void)in_sizes; (void)n_in; (void)out_size; (void)ws_size;
  const float* x       = (const float*)d_in[0];
  const int*   mask    = (const int*)d_in[1];
  const float* bias    = (const float*)d_in[2];
  const float* gamma   = (const float*)d_in[3];
  const float* null_kv = (const float*)d_in[4];
  const float* Wq      = (const float*)d_in[5];
  const float* Wkv     = (const float*)d_in[6];
  const float* Wout    = (const float*)d_in[7];
  float* out = (float*)d_out;

  char* ws = (char*)d_ws;
  size_t off = 0;
  unsigned short* xn    = (unsigned short*)(ws + off); off += (size_t)B_ * N_ * DIM_ * 2;      // 8 MB
  unsigned short* xb    = (unsigned short*)(ws + off); off += (size_t)B_ * N_ * DIM_ * 2;      // 8 MB
  unsigned short* WqT   = (unsigned short*)(ws + off); off += (size_t)HD_ * DIM_ * 2;          // 2 MB
  unsigned short* WkvT  = (unsigned short*)(ws + off); off += (size_t)(2 * DH_) * DIM_ * 2;    // 256 KB
  unsigned short* WoutT = (unsigned short*)(ws + off); off += (size_t)DIM_ * HD_ * 2;          // 2 MB
  unsigned short* qbuf  = (unsigned short*)(ws + off); off += (size_t)B_ * H_ * N_ * DH_ * 2;  // 8 MB
  unsigned short* Kbuf  = (unsigned short*)(ws + off); off += (size_t)B_ * KVROWS * DH_ * 2;   // 520 KB
  unsigned short* Vt    = (unsigned short*)(ws + off); off += (size_t)B_ * DH_ * KVROWS * 2;   // 520 KB
  unsigned short* aout  = (unsigned short*)(ws + off); off += (size_t)B_ * N_ * HD_ * 2;       // 8 MB

  // 1. weights -> bf16, transposed
  transpose_cast_kernel<<<dim3((DIM_ * HD_) / 256), dim3(256), 0, stream>>>(Wq, WqT, DIM_, HD_);
  transpose_cast_kernel<<<dim3((DIM_ * 2 * DH_) / 256), dim3(256), 0, stream>>>(Wkv, WkvT, DIM_, 2 * DH_);
  transpose_cast_kernel<<<dim3((HD_ * DIM_) / 256), dim3(256), 0, stream>>>(Wout, WoutT, HD_, DIM_);
  // 2. layernorm + casts
  layernorm_cast_kernel<<<dim3(B_ * N_), dim3(256), 0, stream>>>(x, gamma, xn, xb);
  // 3. projections (WMMA)
  gemm_q_kernel<<<dim3(HD_ / 64, (B_ * N_) / 128), dim3(256), 0, stream>>>(xn, WqT, qbuf);
  gemm_kv_kernel<<<dim3((2 * DH_) / 64, (B_ * N_) / 128), dim3(256), 0, stream>>>(xb, WkvT, Kbuf, Vt);
  // 4. null tokens + zero padding rows of K/V
  kv_init_kernel<<<dim3((B_ * DH_ * 32) / 256), dim3(256), 0, stream>>>(null_kv, Kbuf, Vt);
  // 5. flash attention (WMMA)
  attention_kernel<<<dim3(N_ / 128, H_, B_), dim3(256), 0, stream>>>(qbuf, Kbuf, Vt, bias, mask, aout);
  // 6. output projection (WMMA)
  gemm_out_kernel<<<dim3(DIM_ / 64, (B_ * N_) / 128), dim3(256), 0, stream>>>(aout, WoutT, out);
}
